// FourierCrossAttention_60146722013305
// MI455X (gfx1250) — matrix operand order repaired and checked by reading, not compiled, not run on hardware
//
#include <hip/hip_runtime.h>

// ---------------- problem constants ----------------
#define BB 16
#define LL 4096
#define HH 8
#define EE 64
#define MM 64   // modes

typedef __attribute__((ext_vector_type(16))) _Float16 v16h;
typedef __attribute__((ext_vector_type(8)))  _Float16 v8h;
typedef __attribute__((ext_vector_type(8)))  float    v8f;

#define DEV __device__ __forceinline__

// D = A(16x32 f16) * B(32x16 f16) + C(16x16 f32), wave32 WMMA
DEV v8f wmma16(v16h a, v16h b, v8f c) {
  return __builtin_amdgcn_wmma_f32_16x16x32_f16(
      /*neg_a=*/false, a, /*neg_b=*/false, b,
      /*c_mod=*/(short)0, c, /*reuse_a=*/false, /*reuse_b=*/false);
}

// A-fragment per-lane load: halves 0..15 at p[0..15]
DEV v16h ldA(const _Float16* p) {
  v8h lo = *(const v8h*)p;
  v8h hi = *(const v8h*)(p + 8);
  return __builtin_shufflevector(lo, hi, 0,1,2,3,4,5,6,7,8,9,10,11,12,13,14,15);
}
// B-fragment per-lane load: 16 contiguous halves (32B)
DEV v16h ldB(const _Float16* p) { return *(const v16h*)p; }

// ---------------------------------------------------------------
// Kernel 1: forward DFT basis, packed in B-fragment order
// basisF[ks(128)][nt(8)][lane(32)][j(16)]; n<64 -> cos, n>=64 -> sin
__global__ __launch_bounds__(256) void k_basisF(_Float16* basisF) {
  unsigned idx = blockIdx.x * 256u + threadIdx.x;          // < 524288
  unsigned j    = idx & 15u;
  unsigned lane = (idx >> 4) & 31u;
  unsigned nt   = (idx >> 9) & 7u;
  unsigned ks   = idx >> 12;
  unsigned t = ks * 32u + ((lane < 16u) ? 0u : 16u) + j;   // 0..4095
  unsigned n = nt * 16u + (lane & 15u);                    // 0..127
  unsigned m = n & 63u;
  unsigned ph = (m * t) & 4095u;                           // exact phase mod L
  float x = (float)ph * (1.0f / 2048.0f);                  // theta/pi
  float v = (n < 64u) ? cospif(x) : sinpif(x);
  basisF[idx] = (_Float16)v;
}

// Kernel 2: inverse DFT basis, packed in B-fragment order
// basisI[ks(4)][nt(256)][lane(32)][j(16)]; k=2m -> cos, k=2m+1 -> sin
__global__ __launch_bounds__(256) void k_basisI(_Float16* basisI) {
  unsigned idx = blockIdx.x * 256u + threadIdx.x;          // < 524288
  unsigned j    = idx & 15u;
  unsigned lane = (idx >> 4) & 31u;
  unsigned nt   = (idx >> 9) & 255u;
  unsigned ks   = idx >> 17;
  unsigned kk = ks * 32u + ((lane < 16u) ? 0u : 16u) + j;  // 0..127
  unsigned t  = nt * 16u + (lane & 15u);                   // 0..4095
  unsigned m  = kk >> 1;
  unsigned ph = (m * t) & 4095u;
  float x = (float)ph * (1.0f / 2048.0f);
  float v = (kk & 1u) ? sinpif(x) : cospif(x);
  basisI[idx] = (_Float16)v;
}

// Kernel 3: repack weights into B-fragment order per (h, mode)
// Wp[h][x][ks(2)][nt(4)][lane(32)][j(16)] ; B[k=e][n=o] = w[h,e,o,x]
__global__ __launch_bounds__(256) void k_wpack(const float* w1, const float* w2,
                                               _Float16* Wp_re, _Float16* Wp_im) {
  unsigned idx = blockIdx.x * 256u + threadIdx.x;          // < 2097152
  unsigned j    = idx & 15u;
  unsigned lane = (idx >> 4) & 31u;
  unsigned nt   = (idx >> 9) & 3u;
  unsigned ks   = (idx >> 11) & 1u;
  unsigned x    = (idx >> 12) & 63u;
  unsigned h    = idx >> 18;
  unsigned e = ks * 32u + ((lane < 16u) ? 0u : 16u) + j;
  unsigned o = nt * 16u + (lane & 15u);
  unsigned src = ((h * 64u + e) * 64u + o) * 64u + x;
  Wp_re[idx] = (_Float16)w1[src];
  Wp_im[idx] = (_Float16)w2[src];
}

// ---------------------------------------------------------------
// Kernel 4: forward truncated DFT for q and k (double-buffered LDS pipeline).
// grid = 256: wg = src(2) x bh(128).  Per wg: C[e=64][n=128] = Q[e][t] * basisF
__global__ __launch_bounds__(256) void k_fdft(
    const float* __restrict__ q, const float* __restrict__ kk,
    const _Float16* __restrict__ basisF,
    _Float16* Xq_re, _Float16* Xq_im,
    _Float16* Xkme_re, _Float16* Xkme_im,
    _Float16* Xkem_re, _Float16* Xkem_im) {
  unsigned wg  = blockIdx.x;
  unsigned src = wg >> 7;            // 0 = q, 1 = k
  unsigned bh  = wg & 127u;
  unsigned b = bh >> 3, h = bh & 7u;
  const float* X = src ? kk : q;

  __shared__ _Float16 As[2][64 * 40];  // [e][t] tiles, stride 40 halves (80B)

  unsigned tid  = threadIdx.x;
  unsigned wave = tid >> 5, lane = tid & 31u;
  unsigned l15 = lane & 15u;
  bool hi = lane >= 16u;

  // cooperative loader: thread covers 8 elements of the 64x32 tile
  unsigned le = tid & 63u;           // e (contiguous in global -> coalesced)
  unsigned lt = tid >> 6;            // t part 0..3 (stride 4)
  const float* gsrc = X + (((unsigned)b * LL) * HH + h) * EE + le;

  v8f acc[4];
#pragma unroll
  for (int mt = 0; mt < 4; ++mt) acc[mt] = (v8f)(0.0f);

  // preload tile 0
#pragma unroll
  for (int i = 0; i < 8; ++i) {
    unsigned tt = lt + (unsigned)i * 4u;
    As[0][le * 40u + tt] = (_Float16)gsrc[(size_t)tt * (HH * EE)];
  }
  __syncthreads();

  for (unsigned ks = 0; ks < 128; ++ks) {
    unsigned cur = ks & 1u;
    // issue next tile's loads while current tile feeds WMMA
    if (ks + 1u < 128u) {
      const float* gn = gsrc + (size_t)(ks + 1u) * 32u * (HH * EE);
#pragma unroll
      for (int i = 0; i < 8; ++i) {
        unsigned tt = lt + (unsigned)i * 4u;
        As[cur ^ 1u][le * 40u + tt] = (_Float16)gn[(size_t)tt * (HH * EE)];
      }
    }
    // B fragment (32B contiguous from packed basis)
    v16h bf = ldB(basisF + (((ks * 8u + wave) * 32u) + lane) * 16u);
#pragma unroll
    for (int mt = 0; mt < 4; ++mt) {
      const _Float16* ab = &As[cur][(mt * 16u + l15) * 40u + (hi ? 16u : 0u)];
      v16h af = ldA(ab);
      acc[mt] = wmma16(af, bf, acc[mt]);
    }
    __syncthreads();
  }

  // epilogue: C row = e, col n = wave*16 + l15.
  // n<64 (wave<4) -> re(m=n); else -> im(m=n-64) = -sum(sin).  Wave-uniform.
  unsigned n = wave * 16u + l15;
  unsigned m = n & 63u;
  bool isRe = (wave < 4u);
  float sgn = isRe ? 1.0f : -1.0f;
  if (src == 0) {
    _Float16* dst = isRe ? Xq_re : Xq_im;
#pragma unroll
    for (int mt = 0; mt < 4; ++mt) {
#pragma unroll
      for (int v = 0; v < 8; ++v) {
        unsigned e = mt * 16u + (unsigned)v + (hi ? 8u : 0u);
        dst[bh * 4096u + m * 64u + e] = (_Float16)(sgn * acc[mt][v]);
      }
    }
  } else {
    _Float16* d1 = isRe ? Xkme_re : Xkme_im;   // [y][e]
    _Float16* d2 = isRe ? Xkem_re : Xkem_im;   // [e][y]
#pragma unroll
    for (int mt = 0; mt < 4; ++mt) {
#pragma unroll
      for (int v = 0; v < 8; ++v) {
        unsigned e = mt * 16u + (unsigned)v + (hi ? 8u : 0u);
        _Float16 hv = (_Float16)(sgn * acc[mt][v]);
        d1[bh * 4096u + m * 64u + e] = hv;
        d2[bh * 4096u + e * 64u + m] = hv;
      }
    }
  }
}

// ---------------------------------------------------------------
// Kernel 5: per (b,h): scores = tanh(Xq^T Xk) (complex), xqkv^T = scores * Xk^T
// grid = 128, block = 256 (8 waves, 2 tiles each of the 4x4 tile grid)
__global__ __launch_bounds__(256) void k_attn(
    const _Float16* __restrict__ Xq_re, const _Float16* __restrict__ Xq_im,
    const _Float16* __restrict__ Xkme_re, const _Float16* __restrict__ Xkme_im,
    const _Float16* __restrict__ Xkem_re, const _Float16* __restrict__ Xkem_im,
    _Float16* xkv_re, _Float16* xkv_im) {
  unsigned bh = blockIdx.x;
  unsigned b = bh >> 3, h = bh & 7u;
  unsigned tid = threadIdx.x;
  unsigned wave = tid >> 5, lane = tid & 31u, l15 = lane & 15u;
  bool hiL = lane >= 16u;
  unsigned koff8  = hiL ? 8u : 0u;
  unsigned koff16 = hiL ? 16u : 0u;

  __shared__ _Float16 Sr[64 * 80];   // scores re, stride 80 halves (160B)
  __shared__ _Float16 Si[64 * 80];

  // GEMM1: scores[x][y] = sum_e Xq[e,x]*Xk[e,y]   (M=x, N=y, K=e)
#pragma unroll
  for (int ti = 0; ti < 2; ++ti) {
    unsigned tl = wave * 2u + (unsigned)ti;
    unsigned mt = tl >> 2, nt = tl & 3u;
    v8f ar = (v8f)(0.0f), ai = (v8f)(0.0f);
#pragma unroll
    for (unsigned ks = 0; ks < 2; ++ks) {
      unsigned aoff = bh * 4096u + (mt * 16u + l15) * 64u + ks * 32u + koff16;
      v16h qr = ldA(Xq_re + aoff);
      v16h qi = ldA(Xq_im + aoff);
      unsigned boff = bh * 4096u + (nt * 16u + l15) * 64u + ks * 32u + koff16;
      v16h kr = ldB(Xkme_re + boff);
      v16h ki = ldB(Xkme_im + boff);
      v16h nqi = -qi;
      ar = wmma16(qr, kr, ar);
      ar = wmma16(nqi, ki, ar);
      ai = wmma16(qr, ki, ai);
      ai = wmma16(qi, kr, ai);
    }
#pragma unroll
    for (int v = 0; v < 8; ++v) {
      unsigned row = mt * 16u + (unsigned)v + (hiL ? 8u : 0u);
      unsigned col = nt * 16u + l15;
      Sr[row * 80u + col] = (_Float16)tanhf(ar[v]);
      Si[row * 80u + col] = (_Float16)tanhf(ai[v]);
    }
  }
  __syncthreads();

  // GEMM2: R[x][e] = sum_y S[x][y]*Xk[e][y]   (M=x, N=e, K=y)
#pragma unroll
  for (int ti = 0; ti < 2; ++ti) {
    unsigned tl = wave * 2u + (unsigned)ti;
    unsigned mt = tl >> 2, nt = tl & 3u;
    v8f rr = (v8f)(0.0f), ri = (v8f)(0.0f);
#pragma unroll
    for (unsigned ks = 0; ks < 2; ++ks) {
      const _Float16* abr = &Sr[(mt * 16u + l15) * 80u + ks * 32u + koff16];
      const _Float16* abi = &Si[(mt * 16u + l15) * 80u + ks * 32u + koff16];
      v16h sr = ldA(abr);
      v16h si = ldA(abi);
      unsigned boff = bh * 4096u + (nt * 16u + l15) * 64u + ks * 32u + koff16;
      v16h k2r = ldB(Xkem_re + boff);
      v16h k2i = ldB(Xkem_im + boff);
      v16h nsi = -si;
      rr = wmma16(sr, k2r, rr);
      rr = wmma16(nsi, k2i, rr);
      ri = wmma16(sr, k2i, ri);
      ri = wmma16(si, k2r, ri);
    }
    // scatter to xkv[h][x][b][e] (f16) for the b-batched weights GEMM
#pragma unroll
    for (int v = 0; v < 8; ++v) {
      unsigned x = mt * 16u + (unsigned)v + (hiL ? 8u : 0u);
      unsigned e = nt * 16u + l15;
      unsigned idx = ((h * 64u + x) * 16u + b) * 64u + e;
      xkv_re[idx] = (_Float16)rr[v];
      xkv_im[idx] = (_Float16)ri[v];
    }
  }
}

// ---------------------------------------------------------------
// Kernel 6: per (h,x): Y[b][o] = sum_e xqkv[b][e] * w[h,e,o,x]  (complex)
// grid = 512, block = 128 (4 waves, wave = ntile). Writes iDFT A-matrix rows.
__global__ __launch_bounds__(128) void k_weights(
    const _Float16* __restrict__ xkv_re, const _Float16* __restrict__ xkv_im,
    const _Float16* __restrict__ Wp_re, const _Float16* __restrict__ Wp_im,
    _Float16* Y_A) {
  unsigned h = blockIdx.x >> 6;
  unsigned x = blockIdx.x & 63u;
  unsigned tid = threadIdx.x;
  unsigned nt = tid >> 5, lane = tid & 31u, l15 = lane & 15u;
  bool hiL = lane >= 16u;
  unsigned koff8 = hiL ? 8u : 0u;

  v8f cr = (v8f)(0.0f), ci = (v8f)(0.0f);
#pragma unroll
  for (unsigned ks = 0; ks < 2; ++ks) {
    unsigned aoff = ((h * 64u + x) * 16u + l15) * 64u + ks * 32u + (hiL ? 16u : 0u);
    v16h arv = ldA(xkv_re + aoff);
    v16h aiv = ldA(xkv_im + aoff);
    unsigned boff = ((((h * 64u + x) * 2u + ks) * 4u + nt) * 32u + lane) * 16u;
    v16h br = ldB(Wp_re + boff);
    v16h bi = ldB(Wp_im + boff);
    v16h nai = -aiv;
    cr = wmma16(arv, br, cr);
    cr = wmma16(nai, bi, cr);
    ci = wmma16(arv, bi, ci);
    ci = wmma16(aiv, br, ci);
  }
  // Y_A[(b*8+h)][o][k] : k=2m -> (m?2:1)*re ; k=2m+1 -> -(2)*im (0 for m=0)
  float s = (x == 0u) ? 1.0f : 2.0f;
#pragma unroll
  for (int v = 0; v < 8; ++v) {
    unsigned b = (unsigned)v + (hiL ? 8u : 0u);
    unsigned o = nt * 16u + l15;
    unsigned base = ((b * 8u + h) * 64u + o) * 128u;
    Y_A[base + 2u * x]      = (_Float16)(s * cr[v]);
    Y_A[base + 2u * x + 1u] = (_Float16)((x == 0u) ? 0.0f : (-s * ci[v]));
  }
}

// ---------------------------------------------------------------
// Kernel 7: per (b,h): out[o][t] = alpha * Y_A[o][:] . basisI[:][t]
// grid = 128, block = 256 (8 waves, each owns 32 N-tiles of 4096 cols)
__global__ __launch_bounds__(256) void k_idft(
    const _Float16* __restrict__ Y_A, const _Float16* __restrict__ basisI,
    float* __restrict__ out) {
  unsigned bh = blockIdx.x;
  unsigned tid = threadIdx.x;
  unsigned wave = tid >> 5, lane = tid & 31u, l15 = lane & 15u;
  bool hiL = lane >= 16u;
  unsigned koff8 = hiL ? 8u : 0u;

  __shared__ _Float16 Ys[64 * 136];   // [o][k], stride 136 halves (272B, 16B-aligned)
#pragma unroll
  for (int i = 0; i < 32; ++i) {
    unsigned idx = tid + (unsigned)i * 256u;   // < 8192
    unsigned o = idx >> 7, kk = idx & 127u;
    Ys[o * 136u + kk] = Y_A[bh * 8192u + idx];
  }
  __syncthreads();

  const float alpha = 1.0f / ((float)LL * 512.0f * 512.0f);  // 2^-30, fp32-applied

  for (unsigned nti = 0; nti < 32; ++nti) {
    unsigned nt = wave * 32u + nti;
    v8f acc[4];
#pragma unroll
    for (int mt = 0; mt < 4; ++mt) acc[mt] = (v8f)(0.0f);
#pragma unroll
    for (unsigned ks = 0; ks < 4; ++ks) {
      v16h bf = ldB(basisI + ((ks * 256u + nt) * 32u + lane) * 16u);
#pragma unroll
      for (int mt = 0; mt < 4; ++mt) {
        const _Float16* ab = &Ys[(mt * 16u + l15) * 136u + ks * 32u + (hiL ? 16u : 0u)];
        v16h af = ldA(ab);
        acc[mt] = wmma16(af, bf, acc[mt]);
      }
    }
#pragma unroll
    for (int mt = 0; mt < 4; ++mt) {
#pragma unroll
      for (int v = 0; v < 8; ++v) {
        unsigned o = mt * 16u + (unsigned)v + (hiL ? 8u : 0u);
        unsigned t = nt * 16u + l15;
        out[((unsigned long)bh * 64u + o) * (unsigned long)LL + t] = alpha * acc[mt][v];
      }
    }
  }
}

// ---------------------------------------------------------------
extern "C" void kernel_launch(void* const* d_in, const int* in_sizes, int n_in,
                              void* d_out, int out_size, void* d_ws, size_t ws_size,
                              hipStream_t stream) {
  (void)in_sizes; (void)n_in; (void)out_size;
  const float* q  = (const float*)d_in[0];
  const float* k  = (const float*)d_in[1];
  // d_in[2] = v (unused by the reference), d_in[5] = mask (unused)
  const float* w1 = (const float*)d_in[3];
  const float* w2 = (const float*)d_in[4];
  float* out = (float*)d_out;

  // workspace carve-up (all f16 halves); total = 20 MiB
  if (ws_size < 21u * 1024u * 1024u) return;
  _Float16* p = (_Float16*)d_ws;
  _Float16* basisF  = p; p += 524288;      // 4096 x 128
  _Float16* basisI  = p; p += 524288;      // 128 x 4096
  _Float16* Wp_re   = p; p += 2097152;     // 8*64 x 64 x 64
  _Float16* Wp_im   = p; p += 2097152;
  _Float16* Xq_re   = p; p += 524288;      // [bh][x][e]
  _Float16* Xq_im   = p; p += 524288;
  _Float16* Xkme_re = p; p += 524288;      // [bh][y][e]
  _Float16* Xkme_im = p; p += 524288;
  _Float16* Xkem_re = p; p += 524288;      // [bh][e][y]
  _Float16* Xkem_im = p; p += 524288;
  _Float16* xkv_re  = p; p += 524288;      // [h][x][b][e]
  _Float16* xkv_im  = p; p += 524288;
  _Float16* Y_A     = p; p += 1048576;     // [bh][o][128]

  k_basisF<<<2048, 256, 0, stream>>>(basisF);
  k_basisI<<<2048, 256, 0, stream>>>(basisI);
  k_wpack <<<8192, 256, 0, stream>>>(w1, w2, Wp_re, Wp_im);
  k_fdft  <<<256, 256, 0, stream>>>(q, k, basisF, Xq_re, Xq_im,
                                    Xkme_re, Xkme_im, Xkem_re, Xkem_im);
  k_attn  <<<128, 256, 0, stream>>>(Xq_re, Xq_im, Xkme_re, Xkme_im,
                                    Xkem_re, Xkem_im, xkv_re, xkv_im);
  k_weights<<<512, 128, 0, stream>>>(xkv_re, xkv_im, Wp_re, Wp_im, Y_A);
  k_idft  <<<128, 256, 0, stream>>>(Y_A, basisI, out);
}
